// Transducer_61040075211263
// MI455X (gfx1250) — matrix operand order repaired and checked by reading, not compiled, not run on hardware
//
#include <hip/hip_runtime.h>
#include <hip/hip_bf16.h>
#include <math.h>

// ---------------- problem constants ----------------
#define B_    2
#define T_    512
#define U_    64
#define UP1   65          // U+1
#define HID_  144
#define CELL_ 320
#define V_    1024
#define VM1   1023        // V-1
#define G4    1280        // 4*CELL
#define MTOT  66560       // B*T*(U+1)
#define BT_   1024        // B*T

typedef _Float16 v16h __attribute__((ext_vector_type(16)));
typedef _Float16 v8h  __attribute__((ext_vector_type(8)));
typedef float    v8f  __attribute__((ext_vector_type(8)));

// ---- TDM availability (probe-verified builtin; arity differs per toolchain) ----
#if defined(__has_builtin)
# if __has_builtin(__builtin_amdgcn_tensor_load_to_lds) && __has_builtin(__builtin_amdgcn_s_wait_tensorcnt)
#  define HAVE_TDM 1
# endif
#endif
#ifndef HAVE_TDM
# define HAVE_TDM 0
#endif

__device__ __forceinline__ float sigm(float x) { return 1.0f / (1.0f + __expf(-x)); }

// ---------------------------------------------------------------------------
// K1: enc[bt,c] = xs[bt,:] . fc0_w[c,:] + fc0_b[c]        (1024 x 320, K=144)
// ---------------------------------------------------------------------------
__global__ void enc_kernel(const float* __restrict__ xs,
                           const float* __restrict__ fc0_w,
                           const float* __restrict__ fc0_b,
                           float* __restrict__ enc) {
    int idx = blockIdx.x * blockDim.x + threadIdx.x;
    if (idx >= BT_ * CELL_) return;
    int bt = idx / CELL_, c = idx % CELL_;
    const float4* xr = (const float4*)(xs + (size_t)bt * HID_);
    const float4* wr = (const float4*)(fc0_w + (size_t)c * HID_);
    float acc = fc0_b[c];
    #pragma unroll 4
    for (int k = 0; k < HID_ / 4; ++k) {
        float4 a = xr[k], w = wr[k];
        acc += a.x * w.x + a.y * w.y + a.z * w.z + a.w * w.w;
    }
    enc[idx] = acc;
}

// ---------------------------------------------------------------------------
// K2: pf[bt,d] = enc[bt,:] . fc1_w[d, 0:CELL]             (1024 x 320, K=320)
// ---------------------------------------------------------------------------
__global__ void pf_kernel(const float* __restrict__ enc,
                          const float* __restrict__ fc1_w,
                          float* __restrict__ pf) {
    int idx = blockIdx.x * blockDim.x + threadIdx.x;
    if (idx >= BT_ * CELL_) return;
    int bt = idx / CELL_, d = idx % CELL_;
    const float4* er = (const float4*)(enc + (size_t)bt * CELL_);
    const float4* wr = (const float4*)(fc1_w + (size_t)d * (2 * CELL_));
    float acc = 0.f;
    #pragma unroll 4
    for (int k = 0; k < CELL_ / 4; ++k) {
        float4 a = er[k], w = wr[k];
        acc += a.x * w.x + a.y * w.y + a.z * w.z + a.w * w.w;
    }
    pf[idx] = acc;
}

// ---------------------------------------------------------------------------
// K3: persistent single-workgroup LSTM over U+1 steps.
// One-hot embedding => input pregate is a gather from w_ih column ys[b,u-1]-1.
// ---------------------------------------------------------------------------
__global__ void lstm_kernel(const int*   __restrict__ ys,
                            const float* __restrict__ w_ih,
                            const float* __restrict__ w_hh,
                            const float* __restrict__ b_ih,
                            const float* __restrict__ b_hh,
                            float* __restrict__ dec) {
    __shared__ __align__(16) float h_lds[B_ * CELL_];
    __shared__ __align__(16) float c_lds[B_ * CELL_];
    __shared__ __align__(16) float g_lds[B_ * G4];
    const int tid = threadIdx.x;
    for (int i = tid; i < B_ * CELL_; i += blockDim.x) { h_lds[i] = 0.f; c_lds[i] = 0.f; }
    __syncthreads();

    for (int u = 0; u < UP1; ++u) {
        for (int o = tid; o < B_ * G4; o += blockDim.x) {
            int b = o / G4, g = o % G4;
            float acc = b_ih[g] + b_hh[g];
            if (u > 0) {
                int yv = ys[b * U_ + (u - 1)];       // in [1, V)
                acc += w_ih[(size_t)g * VM1 + (yv - 1)];
            }
            const float4* wr = (const float4*)(w_hh + (size_t)g * CELL_);
            const float4* hr = (const float4*)(h_lds + b * CELL_);
            #pragma unroll 4
            for (int k = 0; k < CELL_ / 4; ++k) {
                float4 w = wr[k], h = hr[k];
                acc += w.x * h.x + w.y * h.y + w.z * h.z + w.w * h.w;
            }
            g_lds[o] = acc;
        }
        __syncthreads();
        for (int idx = tid; idx < B_ * CELL_; idx += blockDim.x) {
            int b = idx / CELL_, cc = idx % CELL_;
            const float* g = g_lds + b * G4;
            float i_ = sigm(g[cc]);
            float f_ = sigm(g[CELL_ + cc]);
            float gg = tanhf(g[2 * CELL_ + cc]);
            float o_ = sigm(g[3 * CELL_ + cc]);
            float cv = f_ * c_lds[idx] + i_ * gg;
            float hv = o_ * tanhf(cv);
            c_lds[idx] = cv;
            h_lds[idx] = hv;
            dec[((size_t)b * UP1 + u) * CELL_ + cc] = hv;
        }
        __syncthreads();
    }
}

// ---------------------------------------------------------------------------
// K4: pg[b,u,d] = dec[b,u,:] . fc1_w[d, CELL:2*CELL]       (130 x 320, K=320)
// ---------------------------------------------------------------------------
__global__ void pg_kernel(const float* __restrict__ dec,
                          const float* __restrict__ fc1_w,
                          float* __restrict__ pg) {
    int idx = blockIdx.x * blockDim.x + threadIdx.x;
    if (idx >= B_ * UP1 * CELL_) return;
    int bu = idx / CELL_, d = idx % CELL_;
    const float4* er = (const float4*)(dec + (size_t)bu * CELL_);
    const float4* wr = (const float4*)(fc1_w + (size_t)d * (2 * CELL_) + CELL_);
    float acc = 0.f;
    #pragma unroll 4
    for (int k = 0; k < CELL_ / 4; ++k) {
        float4 a = er[k], w = wr[k];
        acc += a.x * w.x + a.y * w.y + a.z * w.z + a.w * w.w;
    }
    pg[idx] = acc;
}

// ---------------------------------------------------------------------------
// K5: pre-swizzle fc2_w (V x CELL f32) into fragment-major f16 for WMMA B.
// Bsw[((tn*10 + ks)*32 + lane)*16 + e]:
//   column v = tn*16 + (lane&15) ; k = ks*32 + (lane>>4)*8 + (e&7) + (e>>3)*16
// => one wave's B fragment is 1KB contiguous; a 4-tile N-group is 40KB flat.
// ---------------------------------------------------------------------------
__global__ void bswz_kernel(const float* __restrict__ fc2_w,
                            _Float16* __restrict__ bsw) {
    int idx = blockIdx.x * blockDim.x + threadIdx.x;
    if (idx >= V_ * CELL_) return;
    int e    = idx & 15;
    int lane = (idx >> 4) & 31;
    int frag = idx >> 9;             // tn*10 + ks
    int ks   = frag % 10;
    int tn   = frag / 10;
    int v    = tn * 16 + (lane & 15);
    int k    = ks * 32 + (lane >> 4) * 8 + (e & 7) + (e >> 3) * 16;
    bsw[idx] = (_Float16)fc2_w[(size_t)v * CELL_ + k];
}

// ---------------------------------------------------------------------------
// TDM 1-D tile load: 40KB contiguous global -> LDS via Tensor Data Mover.
// D# per CDNA5 ISA: group0 {count=1, lds_addr, global_addr,
// type=2}; group1 {data_size=8B, tensor_dim0=tile_dim0=N8, stride=N8}.
// ---------------------------------------------------------------------------
#if HAVE_TDM
typedef unsigned int v4u  __attribute__((ext_vector_type(4)));
typedef int          v8i_ __attribute__((ext_vector_type(8)));
typedef int          v4i_ __attribute__((ext_vector_type(4)));

__device__ __forceinline__ void tdm_load_1d(const void* gptr, unsigned lds_off,
                                            unsigned n8 /* bytes/8 */) {
    unsigned long long ga = (unsigned long long)(uintptr_t)gptr;
    v4u g0;
    g0[0] = 1u;                                           // count=1, user mode
    g0[1] = lds_off;                                      // lds_addr (bytes)
    g0[2] = (unsigned)(ga & 0xffffffffu);                 // global_addr[31:0]
    g0[3] = (unsigned)((ga >> 32) & 0x1ffffffu) | (2u << 30); // [56:32] | type=2
    v8i_ g1;
    g1[0] = (3 << 16);                                    // mask=0, data_size=8B
    g1[1] = (int)((n8 & 0xffffu) << 16);                  // tensor_dim0 lo16
    g1[2] = (int)(((n8 >> 16) & 0xffffu) | (1u << 16));   // dim0 hi16 | dim1 lo16=1
    g1[3] = (int)((n8 & 0xffffu) << 16);                  // dim1 hi=0 | tile_dim0=n8
    g1[4] = 1;                                            // tile_dim1=1, tile_dim2=0
    g1[5] = (int)n8;                                      // dim0_stride lo32
    g1[6] = 0;
    g1[7] = 0;
    v4i_ z4 = {0, 0, 0, 0};
#if __clang_major__ >= 23
    v8i_ z8 = {0, 0, 0, 0, 0, 0, 0, 0};
    __builtin_amdgcn_tensor_load_to_lds(g0, g1, z4, z4, z8, 0);
#else
    __builtin_amdgcn_tensor_load_to_lds(g0, g1, z4, z4, 0);
#endif
}
#endif

// ---------------------------------------------------------------------------
// K6: joint + fc2 via f16 WMMA, B double-buffered through LDS (TDM async).
// Block: 256 thr (8 waves), 64 M rows x full N=1024, K=320 (10 x 32).
// N processed in 16 groups of 64 cols (4 tiles); each wave: 1 M-strip x 2 tiles
// per group (two independent accumulator chains for WMMA ILP).
// NOTE: no local arrays of LDS-derived pointers (addrspacecast constants are
// not linkable); buffer pointers are computed arithmetically from `buf`.
// ---------------------------------------------------------------------------
#define A_STRIDE   328                    // f16 elems; pad breaks bank stride
#define A_BYTES    (64 * A_STRIDE * 2)    // 41984, 128B-aligned
#define GRP_ELEMS  (4 * 10 * 512)         // 20480 f16 per 64-col group
#define GRP_BYTES  (GRP_ELEMS * 2)        // 40960
#define SMEM_BYTES (A_BYTES + 2 * GRP_BYTES)

__global__ void joint_kernel(const float* __restrict__ pf,
                             const float* __restrict__ pg,
                             const float* __restrict__ fc1_b,
                             const _Float16* __restrict__ bsw,
                             const float* __restrict__ fc2_b,
                             float* __restrict__ out) {
    extern __shared__ __align__(128) char smem[];
    _Float16* lds_a = (_Float16*)smem;
    char*     bbase = smem + A_BYTES;     // two GRP_BYTES buffers follow A

    const int m0 = blockIdx.x * 64;

    // ---- phase 1: A tile = f16(tanh(pf + pg + fc1_b)), 64 x 320 ----
    for (int idx = threadIdx.x; idx < 64 * CELL_; idx += blockDim.x) {
        int r = idx / CELL_, d = idx % CELL_;
        int m = m0 + r;
        int b   = m / (T_ * UP1);
        int rem = m % (T_ * UP1);
        int t = rem / UP1;
        int u = rem % UP1;
        float val = pf[((size_t)b * T_ + t) * CELL_ + d]
                  + pg[((size_t)b * UP1 + u) * CELL_ + d]
                  + fc1_b[d];
        lds_a[r * A_STRIDE + d] = (_Float16)tanhf(val);
    }

#if HAVE_TDM
    // prefetch group 0 (one TDM issue from wave 0; EXEC-independent DMA)
    if (threadIdx.x < 32)
        tdm_load_1d(bsw, (unsigned)(uintptr_t)bbase, GRP_BYTES / 8);
#endif
    __syncthreads();

    const int wave  = threadIdx.x >> 5;
    const int lane  = threadIdx.x & 31;
    const int strip = wave & 3;           // 16-row M strip
    const int tnsel = wave >> 2;          // which pair of the group's 4 tiles
    const int half  = lane >> 4;
    const int l15   = lane & 15;

    // A fragments, all K, kept in registers (80 VGPRs) for the whole kernel.
    // 16-bit A layout (16x32): lanes 0-15: K 0-7 & 16-23; lanes 16-31: K 8-15 & 24-31.
    const _Float16* arow = &lds_a[(strip * 16 + l15) * A_STRIDE];
    v16h afrag[10];
    #pragma unroll
    for (int ks = 0; ks < 10; ++ks) {
        const int k0 = ks * 32;
        v8h alo = *(const v8h*)(arow + k0 + half * 8);
        v8h ahi = *(const v8h*)(arow + k0 + 16 + half * 8);
        #pragma unroll
        for (int e = 0; e < 8; ++e) { afrag[ks][e] = alo[e]; afrag[ks][e + 8] = ahi[e]; }
    }

    for (int g = 0; g < 16; ++g) {
#if HAVE_TDM
        const int buf = g & 1;
        if (threadIdx.x < 32) {
            if (g + 1 < 16)
                tdm_load_1d(bsw + (size_t)(g + 1) * GRP_ELEMS,
                            (unsigned)(uintptr_t)(bbase + (size_t)((g + 1) & 1) * GRP_BYTES),
                            GRP_BYTES / 8);
        }
        if (g + 1 < 16) __builtin_amdgcn_s_wait_tensorcnt(1);  // group g done
        else            __builtin_amdgcn_s_wait_tensorcnt(0);  // last: drain
        __syncthreads();
#else
        const int buf = 0;
        {   // cooperative synchronous copy of group g (fallback path)
            const v16h* src = (const v16h*)(bsw + (size_t)g * GRP_ELEMS);
            v16h*       dst = (v16h*)bbase;
            for (int i = threadIdx.x; i < GRP_ELEMS / 16; i += blockDim.x)
                dst[i] = src[i];
        }
        __syncthreads();
#endif
        const _Float16* bb = (const _Float16*)(bbase + (size_t)buf * GRP_BYTES);

        // two tiles per wave: local tn = tnsel*2 + {0,1}
        const int tnl0 = tnsel * 2;
        v8f acc0 = {}, acc1 = {};
        #pragma unroll
        for (int ks = 0; ks < 10; ++ks) {
            v16h b0 = *(const v16h*)(bb + ((tnl0 * 10 + ks) * 32 + lane) * 16);
            v16h b1 = *(const v16h*)(bb + (((tnl0 + 1) * 10 + ks) * 32 + lane) * 16);
            acc0 = __builtin_amdgcn_wmma_f32_16x16x32_f16(
                false, afrag[ks], false, b0, (short)0, acc0, false, false);
            acc1 = __builtin_amdgcn_wmma_f32_16x16x32_f16(
                false, afrag[ks], false, b1, (short)0, acc1, false, false);
        }
        // C/D layout: VGPR j -> (M = half*8 + j, N = l15)
        const int   mrow = m0 + strip * 16 + half * 8;
        const int   tng  = g * 4 + tnl0;
        const int   vc0  = tng * 16 + l15;
        const int   vc1  = (tng + 1) * 16 + l15;
        const float bia0 = fc2_b[vc0];
        const float bia1 = fc2_b[vc1];
        size_t base0 = (size_t)mrow * V_ + vc0;
        size_t base1 = (size_t)mrow * V_ + vc1;
        #pragma unroll
        for (int j = 0; j < 8; ++j) {
            out[base0 + (size_t)j * V_] = acc0[j] + bia0;
            out[base1 + (size_t)j * V_] = acc1[j] + bia1;
        }
        __syncthreads();   // group buffer reuse fence
    }
}

// ---------------------------------------------------------------------------
// launch
// ---------------------------------------------------------------------------
extern "C" void kernel_launch(void* const* d_in, const int* in_sizes, int n_in,
                              void* d_out, int out_size, void* d_ws, size_t ws_size,
                              hipStream_t stream) {
    (void)in_sizes; (void)n_in; (void)out_size; (void)ws_size;
    const float* xs    = (const float*)d_in[0];
    const int*   ys    = (const int*)  d_in[1];
    const float* fc0_w = (const float*)d_in[2];
    const float* fc0_b = (const float*)d_in[3];
    const float* w_ih  = (const float*)d_in[4];
    const float* w_hh  = (const float*)d_in[5];
    const float* b_ih  = (const float*)d_in[6];
    const float* b_hh  = (const float*)d_in[7];
    const float* fc1_w = (const float*)d_in[8];
    const float* fc1_b = (const float*)d_in[9];
    const float* fc2_w = (const float*)d_in[10];
    const float* fc2_b = (const float*)d_in[11];
    float* out = (float*)d_out;

    char* ws = (char*)d_ws;
    float*    pf  = (float*)(ws);                          // 1024*320 f32
    float*    pg  = (float*)(ws + 1310720);                //  130*320 f32
    float*    dec = (float*)(ws + 1477120);                //  130*320 f32
    float*    enc = (float*)(ws + 1643520);                // 1024*320 f32
    _Float16* bsw = (_Float16*)(ws + 2954240);             // 1024*320 f16 swizzled

    enc_kernel<<<(BT_ * CELL_ + 255) / 256, 256, 0, stream>>>(xs, fc0_w, fc0_b, enc);
    pf_kernel <<<(BT_ * CELL_ + 255) / 256, 256, 0, stream>>>(enc, fc1_w, pf);
    bswz_kernel<<<(V_ * CELL_ + 255) / 256, 256, 0, stream>>>(fc2_w, bsw);
    lstm_kernel<<<1, 1024, 0, stream>>>(ys, w_ih, w_hh, b_ih, b_hh, dec);
    pg_kernel <<<(B_ * UP1 * CELL_ + 255) / 256, 256, 0, stream>>>(dec, fc1_w, pg);
    joint_kernel<<<MTOT / 64, 256, SMEM_BYTES, stream>>>(pf, pg, fc1_b, bsw, fc2_b, out);
}
